// RoPELinearAttention_28467043238017
// MI455X (gfx1250) — matrix operand order repaired and checked by reading, not compiled, not run on hardware
//
#include <hip/hip_runtime.h>

typedef __attribute__((ext_vector_type(16))) _Float16 v16h;
typedef __attribute__((ext_vector_type(8)))  float    v8f;

#define NB    8     // batch
#define NHD   8     // heads
#define DH    32    // head dim
#define CHUNK 256   // sequence rows per block
#define SP    (CHUNK / 2)  // K-pairs per chunk (kernel 2)
#define SPAD  132   // padded row stride (dwords) for [channel][s-pair] tiles
#define DPAD  20    // padded row stride (dwords) for [row][d-pair] tiles
#define EPS   1e-6f
#define INV_LN 0.14391156f   // ln(10000)/64

union frag16 { v16h h; unsigned int u[8]; };

__device__ __forceinline__ float elu1(float x) {
    // elu(x)+1 : x>0 -> x+1 ; x<=0 -> exp(x)
    return x > 0.0f ? x + 1.0f : __expf(x);
}

__device__ __forceinline__ unsigned int pkh2(float a, float b) {
    union { _Float16 f[2]; unsigned int u; } p;
    p.f[0] = (_Float16)a; p.f[1] = (_Float16)b;
    return p.u;
}

// Interleaved RoPE on a channel pair (d=2*dp, d=2*dp+1) held by one thread.
// One sincos per pair: q and t are identical for both channels of a pair.
__device__ __forceinline__ void rope2(int l, int W, int head, int dp,
                                      float& xe, float& xo) {
    int i = l / W + 1;
    int j = l - (l / W) * W + 1;
    int q = head * 8 + (dp >> 1);        // global freq index in [0,64)
    float pos = (dp & 1) ? (float)j : (float)i;
    float s, c;
    __sincosf(pos * __expf(-(float)q * INV_LN), &s, &c);
    float re = xe * c - xo * s;
    float ro = xo * c + xe * s;
    xe = re; xo = ro;
}

// ---------------------------------------------------------------------------
// Kernel 1: Ksum[n,h,d] = sum_s elu1(K[n,s,h,d]) * kv_mask[n,s]
// ---------------------------------------------------------------------------
__global__ __launch_bounds__(256)
void ksum_kernel(const float* __restrict__ keys,
                 const unsigned char* __restrict__ kv_mask,
                 float* __restrict__ ksum, int L) {
    const int b = blockIdx.x;
    const int n = b / NHD, h = b % NHD;
    const int tid = threadIdx.x, lane = tid & 31, wave = tid >> 5;

    float acc = 0.0f;
    const float* kp = keys + (((size_t)n * L) * NHD + h) * DH + lane;
    const unsigned char* mp = kv_mask + (size_t)n * L;
    for (int s = wave; s < L; s += 8) {
        float kx = kp[(size_t)s * NHD * DH];
        float m  = mp[s] ? 1.0f : 0.0f;
        acc += elu1(kx) * m;
    }
    __shared__ float red[256];
    red[tid] = acc;
    __syncthreads();
    if (tid < DH) {
        float sum = 0.0f;
        #pragma unroll
        for (int w = 0; w < 8; ++w) sum += red[w * 32 + tid];
        ksum[(n * NHD + h) * DH + tid] = sum;
    }
}

// ---------------------------------------------------------------------------
// Kernel 2: KV[n,h,d,v] += sum_{s in chunk} Kr[n,s,h,d] * V[n,s,h,v]
// Staging LDS layout: [channel][s-pair] as packed f16x2 dwords, so WMMA
// fragment registers are direct (vectorizable) b32/b128 LDS loads.
// ---------------------------------------------------------------------------
__global__ __launch_bounds__(256)
void kv_kernel(const float* __restrict__ keys,
               const float* __restrict__ values,
               const unsigned char* __restrict__ kv_mask,
               const int* __restrict__ dW,
               float* __restrict__ kv, int L) {
    const int n = blockIdx.z, h = blockIdx.y;
    const int s0 = blockIdx.x * CHUNK;
    const int W = *dW;
    const int tid = threadIdx.x, lane = tid & 31, wave = tid >> 5;

    __shared__ unsigned int krsT[DH * SPAD];   // Kr, [channel d][s-pair]
    __shared__ unsigned int vssT[DH * SPAD];   // V , [channel v][s-pair]
    __shared__ float        red[8 * DH * DH];  // per-wave partial KV

    const size_t nL = (size_t)n * L;

    // Phase 1: stream + transform. Thread owns channel pair (2dp, 2dp+1) and
    // both rows of one of the wave's two s-pairs; packs f16x2 along s.
    {
        const int dp = lane & 15;
        const int rh = lane >> 4;
        for (int rpb = wave * 2; rpb < SP; rpb += 16) {
            int rp = rpb + rh;                 // s-pair index in chunk
            int sA = s0 + 2 * rp;              // rows sA, sA+1
            int scA = sA     < L ? sA     : L - 1;
            int scB = sA + 1 < L ? sA + 1 : L - 1;
            float mA = (sA     < L && kv_mask[nL + scA]) ? 1.0f : 0.0f;
            float mB = (sA + 1 < L && kv_mask[nL + scB]) ? 1.0f : 0.0f;
            size_t baseA = ((nL + scA) * NHD + h) * DH + 2 * dp;
            size_t baseB = ((nL + scB) * NHD + h) * DH + 2 * dp;
            float2 kA = *(const float2*)(keys   + baseA);
            float2 kB = *(const float2*)(keys   + baseB);
            float2 vA = *(const float2*)(values + baseA);
            float2 vB = *(const float2*)(values + baseB);
            float keA = elu1(kA.x) * mA, koA = elu1(kA.y) * mA;
            float keB = elu1(kB.x) * mB, koB = elu1(kB.y) * mB;
            rope2(scA, W, h, dp, keA, koA);
            rope2(scB, W, h, dp, keB, koB);
            krsT[(2 * dp)     * SPAD + rp] = pkh2(keA, keB);
            krsT[(2 * dp + 1) * SPAD + rp] = pkh2(koA, koB);
            vssT[(2 * dp)     * SPAD + rp] = pkh2(vA.x * mA, vB.x * mB);
            vssT[(2 * dp + 1) * SPAD + rp] = pkh2(vA.y * mA, vB.y * mB);
        }
    }
    __syncthreads();

    // Phase 2: each wave contracts its 32-row slice (one K=32 WMMA step).
    // A = Kr^T (M=d, K=s), B = V (K=s, N=v).
    const int slp = wave * 16;                 // s-pair base for this wave
    frag16 A0, A1, B0, B1;
    {
        int row = lane & 15;
        int kbp = (lane >= 16) ? 8 : 0;        // A K-pair offset
        #pragma unroll
        for (int v = 0; v < 8; ++v) {
            int sp = slp + kbp + v;
            A0.u[v] = krsT[row        * SPAD + sp];
            A1.u[v] = krsT[(16 + row) * SPAD + sp];
        }
        int col  = lane & 15;
        int kb2p = (lane >= 16) ? 8 : 0;       // B K-pair offset
        #pragma unroll
        for (int j = 0; j < 8; ++j) {
            B0.u[j] = vssT[col        * SPAD + slp + kb2p + j];
            B1.u[j] = vssT[(16 + col) * SPAD + slp + kb2p + j];
        }
    }
    v8f c00 = {}, c01 = {}, c10 = {}, c11 = {};
    c00 = __builtin_amdgcn_wmma_f32_16x16x32_f16(false, A0.h, false, B0.h, (short)0, c00, false, false);
    c01 = __builtin_amdgcn_wmma_f32_16x16x32_f16(false, A0.h, false, B1.h, (short)0, c01, false, false);
    c10 = __builtin_amdgcn_wmma_f32_16x16x32_f16(false, A1.h, false, B0.h, (short)0, c10, false, false);
    c11 = __builtin_amdgcn_wmma_f32_16x16x32_f16(false, A1.h, false, B1.h, (short)0, c11, false, false);

    // Scatter fragments to per-wave LDS (C layout: M=r+8*(lane>=16), N=lane&15).
    {
        int mh = (lane >= 16) ? 8 : 0;
        int nn = lane & 15;
        float* rw = red + wave * (DH * DH);
        #pragma unroll
        for (int r = 0; r < 8; ++r) {
            rw[(r + mh)      * DH + nn]      = c00[r];
            rw[(r + mh)      * DH + 16 + nn] = c01[r];
            rw[(16 + r + mh) * DH + nn]      = c10[r];
            rw[(16 + r + mh) * DH + 16 + nn] = c11[r];
        }
    }
    __syncthreads();

    // Reduce 8 waves, split-K accumulate into global KV.
    float* kvp = kv + ((size_t)n * NHD + h) * (DH * DH);
    for (int e = tid; e < DH * DH; e += 256) {
        float sum = 0.0f;
        #pragma unroll
        for (int w = 0; w < 8; ++w) sum += red[w * DH * DH + e];
        atomicAdd(&kvp[e], sum);
    }
}

// ---------------------------------------------------------------------------
// Kernel 3: out[n,l,h,v] = Z[n,l,h] * (Qr[n,l,h,:] @ KV[n,h,:,:])
// Z = 1/(Qelu . Ksum + eps).  (v/L and *L from the reference cancel.)
// Staging LDS: Qr as [row][d-pair], KV as [v][d-pair] packed f16x2 dwords.
// ---------------------------------------------------------------------------
__global__ __launch_bounds__(256)
void out_kernel(const float* __restrict__ queries,
                const unsigned char* __restrict__ q_mask,
                const float* __restrict__ ksum,
                const float* __restrict__ kv,
                const int* __restrict__ dW,
                float* __restrict__ out, int L) {
    const int n = blockIdx.z, h = blockIdx.y;
    const int l0 = blockIdx.x * CHUNK;
    const int W = *dW;
    const int tid = threadIdx.x, lane = tid & 31, wave = tid >> 5;

    __shared__ unsigned int qrsP[CHUNK * DPAD];  // Qr, [row][d-pair]
    __shared__ unsigned int kvhT[DH * DPAD];     // KV, [v][d-pair]
    __shared__ float        zbuf[CHUNK];
    __shared__ float        ks[DH];

    const size_t nL = (size_t)n * L;
    const float* kvp = kv + ((size_t)n * NHD + h) * (DH * DH);

    // KV (32x32 f32) -> packed f16x2, transposed so K(=d) pairs are contiguous.
    for (int e = tid; e < DH * DH / 2; e += 256) {
        int dp = e & 15, vd = e >> 4;
        kvhT[vd * DPAD + dp] = pkh2(kvp[(2 * dp) * DH + vd],
                                    kvp[(2 * dp + 1) * DH + vd]);
    }
    if (tid < DH) ks[tid] = ksum[(n * NHD + h) * DH + tid];
    __syncthreads();

    // Phase 1: stream Q rows, compute Z, RoPE, stage packed f16 Qr.
    {
        const int dp = lane & 15;
        const int rh = lane >> 4;
        for (int rb = wave * 2 + rh; rb < CHUNK; rb += 16) {
            int l  = l0 + rb;
            int lc = l < L ? l : L - 1;
            float m = (l < L && q_mask[nL + lc]) ? 1.0f : 0.0f;
            size_t base = ((nL + lc) * NHD + h) * DH + 2 * dp;
            float2 q2 = *(const float2*)(queries + base);
            float xe = elu1(q2.x) * m, xo = elu1(q2.y) * m;
            float p = xe * ks[2 * dp] + xo * ks[2 * dp + 1];
            #pragma unroll
            for (int off = 8; off > 0; off >>= 1) p += __shfl_xor(p, off);
            if (dp == 0) zbuf[rb] = 1.0f / (p + EPS);
            rope2(lc, W, h, dp, xe, xo);
            qrsP[rb * DPAD + dp] = pkh2(xe, xo);
        }
    }
    __syncthreads();

    // Phase 2: each wave does a 32x32 GEMM tile (K = DH = 32).
    // A = Qr (M=row, K=d), B = KV (K=d, N=v).
    const int m0 = wave * 32;
    frag16 A0, A1, B0, B1;
    {
        int row = lane & 15;
        int kbp = (lane >= 16) ? 8 : 0;
        #pragma unroll
        for (int v = 0; v < 8; ++v) {
            int kp = kbp + v;
            A0.u[v] = qrsP[(m0 + row)      * DPAD + kp];
            A1.u[v] = qrsP[(m0 + 16 + row) * DPAD + kp];
        }
        int col  = lane & 15;
        int kb2p = (lane >= 16) ? 8 : 0;
        #pragma unroll
        for (int j = 0; j < 8; ++j) {
            B0.u[j] = kvhT[col        * DPAD + kb2p + j];
            B1.u[j] = kvhT[(16 + col) * DPAD + kb2p + j];
        }
    }
    v8f c00 = {}, c01 = {}, c10 = {}, c11 = {};
    c00 = __builtin_amdgcn_wmma_f32_16x16x32_f16(false, A0.h, false, B0.h, (short)0, c00, false, false);
    c01 = __builtin_amdgcn_wmma_f32_16x16x32_f16(false, A0.h, false, B1.h, (short)0, c01, false, false);
    c10 = __builtin_amdgcn_wmma_f32_16x16x32_f16(false, A1.h, false, B0.h, (short)0, c10, false, false);
    c11 = __builtin_amdgcn_wmma_f32_16x16x32_f16(false, A1.h, false, B1.h, (short)0, c11, false, false);

    // Scale by Z and write out.
    const int mh = (lane >= 16) ? 8 : 0;
    const int nn = lane & 15;
    #pragma unroll
    for (int r = 0; r < 8; ++r) {
        int ml0 = m0 + r + mh;        // rows of tiles (0,*)
        int ml1 = m0 + 16 + r + mh;   // rows of tiles (1,*)
        int la = l0 + ml0, lb = l0 + ml1;
        if (la < L) {
            float z = zbuf[ml0];
            size_t ob = ((nL + la) * NHD + h) * DH;
            out[ob + nn]      = c00[r] * z;
            out[ob + 16 + nn] = c01[r] * z;
        }
        if (lb < L) {
            float z = zbuf[ml1];
            size_t ob = ((nL + lb) * NHD + h) * DH;
            out[ob + nn]      = c10[r] * z;
            out[ob + 16 + nn] = c11[r] * z;
        }
    }
}

// ---------------------------------------------------------------------------
extern "C" void kernel_launch(void* const* d_in, const int* in_sizes, int n_in,
                              void* d_out, int out_size, void* d_ws, size_t ws_size,
                              hipStream_t stream) {
    const float* queries = (const float*)d_in[0];
    const float* keys    = (const float*)d_in[1];
    const float* values  = (const float*)d_in[2];
    const unsigned char* q_mask  = (const unsigned char*)d_in[3];
    const unsigned char* kv_mask = (const unsigned char*)d_in[4];
    const int* dW = (const int*)d_in[6];   // d_in[5] = H (unused directly)

    const int L = in_sizes[0] / (NB * NHD * DH);

    float* ksum = (float*)d_ws;                    // NB*NH*DH floats
    float* kvws = ksum + NB * NHD * DH;            // NB*NH*DH*DH floats

    hipMemsetAsync(kvws, 0, sizeof(float) * NB * NHD * DH * DH, stream);

    ksum_kernel<<<dim3(NB * NHD), 256, 0, stream>>>(keys, kv_mask, ksum, L);

    const int chunks = (L + CHUNK - 1) / CHUNK;
    kv_kernel<<<dim3(chunks, NHD, NB), 256, 0, stream>>>(keys, values, kv_mask, dW, kvws, L);
    out_kernel<<<dim3(chunks, NHD, NB), 256, 0, stream>>>(queries, q_mask, ksum, kvws, dW,
                                                          (float*)d_out, L);
}
